// GoEModel_74199855006274
// MI455X (gfx1250) — matrix operand order repaired and checked by reading, not compiled, not run on hardware
//
#include <hip/hip_runtime.h>

// ----- model dims -----
constexpr int cV = 32000, cD = 512, cE = 4, cNH = 8, cHD = 64, cL = 2,
              cFF = 2048, cRH = 256;
constexpr int cB = 8, cS = 256, cT = cB * cS, cMPL = 3, cMV = 2;

typedef __attribute__((ext_vector_type(16))) _Float16 v16h;
typedef __attribute__((ext_vector_type(8)))  float    v8f;
typedef __attribute__((ext_vector_type(4)))  float    fx4;
typedef __attribute__((ext_vector_type(4)))  _Float16 hx4;

__device__ __forceinline__ float gelu_f(float v) {
  return 0.5f * v * (1.0f + erff(v * 0.7071067811865475f));
}

// blockDim.x == 256 assumed
__device__ __forceinline__ float block_sum(float v, float* red) {
  int tid = threadIdx.x;
  red[tid] = v; __syncthreads();
  for (int s = 128; s > 0; s >>= 1) {
    if (tid < s) red[tid] += red[tid + s];
    __syncthreads();
  }
  float r = red[0]; __syncthreads();
  return r;
}

// ================= WMMA GEMM =================
// C[M,N] = act( A[M,K] * op(B) + bias + res )
// gemm_wnk: B given as W[N,K] (i.e. C = A * W^T)  -- all linear layers, Q*K^T
// gemm_bkn: B given row-major [K,N]               -- P*V
// Batch (grid.z) decomposed as z = b*zDiv + h ; per-operand strides for b and h.
// Expert indirection: if sel != nullptr, e = sel[b]; e < 0 => skip block;
// weight/bias get + e*stride.
struct GemmP {
  const float* A; long lda, aStrideB, aStrideH;
  const float* W; long ldw, wStrideB, wStrideH, wStrideSel, kOff;
  const float* bias; long biasStrideSel;
  const float* res;                       // same geometry as C (may alias C)
  float* C; long ldc, cStrideB, cStrideH;
  int K;                                  // reduction length (multiple of 64)
  int act;                                // 0 none, 1 gelu, 2 sigmoid
  int zDiv;
  const int* sel;
};

#define BM 64
#define BN 64
#define BKK 64
#define APAD 8

__global__ __launch_bounds__(256) void gemm_wnk(GemmP p) {
  __shared__ _Float16 As[BM][BKK + APAD];
  __shared__ _Float16 Ws[BN][BKK + APAD];
  const int z = blockIdx.z;
  const int bidx = z / p.zDiv, hidx = z % p.zDiv;
  long wOff = (long)bidx * p.wStrideB + (long)hidx * p.wStrideH + p.kOff;
  long bOffB = 0;
  if (p.sel) {
    int e = p.sel[bidx];
    if (e < 0) return;
    wOff += (long)e * p.wStrideSel;
    bOffB = (long)e * p.biasStrideSel;
  }
  const long aOff = (long)bidx * p.aStrideB + (long)hidx * p.aStrideH;
  const long cOff = (long)bidx * p.cStrideB + (long)hidx * p.cStrideH;
  const int tid = threadIdx.x;
  const int lane = tid & 31, wave = tid >> 5;
  const int wm = wave & 3, wn = wave >> 2;         // 4 x 2 wave grid
  const long rowBase = (long)blockIdx.y * BM;
  const long colBase = (long)blockIdx.x * BN;
  v8f acc[2] = {};
  for (int k0 = 0; k0 < p.K; k0 += BKK) {
    // ---- stage 64x64 fp32 tiles -> f16 LDS, vectorized b128 loads ----
    fx4 va[4], vw[4];
#pragma unroll
    for (int j = 0; j < 4; ++j) {
      int idx = tid + 256 * j;                     // 1024 float4s per tile
      int r = idx >> 4, c = (idx & 15) << 2;
      va[j] = *(const fx4*)(p.A + aOff + (rowBase + r) * p.lda + k0 + c);
      vw[j] = *(const fx4*)(p.W + wOff + (colBase + r) * p.ldw + k0 + c);
    }
#pragma unroll
    for (int j = 0; j < 4; ++j) {
      int idx = tid + 256 * j;
      int r = idx >> 4, c = (idx & 15) << 2;
      hx4 ha, hw;
#pragma unroll
      for (int q = 0; q < 4; ++q) {
        ha[q] = (_Float16)va[j][q];
        hw[q] = (_Float16)vw[j][q];
      }
      *(hx4*)(&As[r][c]) = ha;
      *(hx4*)(&Ws[r][c]) = hw;
    }
    __syncthreads();
    if (k0 + BKK < p.K) {
      __builtin_prefetch(p.A + aOff + (rowBase + (tid >> 2)) * p.lda + k0 + BKK, 0, 1);
      __builtin_prefetch(p.W + wOff + (colBase + (tid >> 2)) * p.ldw + k0 + BKK, 0, 1);
    }
    const int ar = wm * 16 + (lane & 15);
    const int aoff = (lane >> 4) * 8;
    const int n0 = wn * 32 + (lane & 15);
    const int koff = (lane >> 4) * 16;
#pragma unroll
    for (int ks = 0; ks < 2; ++ks) {
      // A fragment: 16x32, ISA layout (lanes 0-15: K 0-7/16-23; 16-31: 8-15/24-31)
      v16h a, b0, b1;
#pragma unroll
      for (int j = 0; j < 16; ++j) {
        int kk = ks * 32 + j + aoff + ((j >= 8) ? 8 : 0);
        a[j] = As[ar][kk];
      }
      // B fragment: B[k][n] = W[n][k]; same K layout as A (lanes 0-15: K 0-7/16-23; 16-31: 8-15/24-31)
#pragma unroll
      for (int j = 0; j < 16; ++j) {
        b0[j] = Ws[n0][ks * 32 + aoff + j + ((j >= 8) ? 8 : 0)];
        b1[j] = Ws[n0 + 16][ks * 32 + aoff + j + ((j >= 8) ? 8 : 0)];
      }
      acc[0] = __builtin_amdgcn_wmma_f32_16x16x32_f16(false, a, false, b0,
                                                      (short)0, acc[0], false, false);
      acc[1] = __builtin_amdgcn_wmma_f32_16x16x32_f16(false, a, false, b1,
                                                      (short)0, acc[1], false, false);
    }
    __syncthreads();
  }
  // C layout: VGPR i -> M = i (+8 for lanes 16-31), N = lane&15
  const int rsub = wm * 16 + ((lane >> 4) << 3);
  const int csub = wn * 32 + (lane & 15);
#pragma unroll
  for (int t = 0; t < 2; ++t) {
    long col = colBase + csub + t * 16;
    float bv = p.bias ? p.bias[bOffB + col] : 0.0f;
#pragma unroll
    for (int i = 0; i < 8; ++i) {
      long row = rowBase + rsub + i;
      float v = acc[t][i] + bv;
      if (p.res) v += p.res[cOff + row * p.ldc + col];
      if (p.act == 1) v = gelu_f(v);
      else if (p.act == 2) v = 1.0f / (1.0f + expf(-v));
      p.C[cOff + row * p.ldc + col] = v;
    }
  }
}

__global__ __launch_bounds__(256) void gemm_bkn(GemmP p) {
  __shared__ _Float16 As[BM][BKK + APAD];
  __shared__ _Float16 Bs[BKK][BN + APAD];
  const int z = blockIdx.z;
  const int bidx = z / p.zDiv, hidx = z % p.zDiv;
  if (p.sel && p.sel[bidx] < 0) return;
  const long aOff = (long)bidx * p.aStrideB + (long)hidx * p.aStrideH;
  const long bOff = (long)bidx * p.wStrideB + (long)hidx * p.wStrideH + p.kOff;
  const long cOff = (long)bidx * p.cStrideB + (long)hidx * p.cStrideH;
  const int tid = threadIdx.x;
  const int lane = tid & 31, wave = tid >> 5;
  const int wm = wave & 3, wn = wave >> 2;
  const long rowBase = (long)blockIdx.y * BM;
  const long colBase = (long)blockIdx.x * BN;
  v8f acc[2] = {};
  for (int k0 = 0; k0 < p.K; k0 += BKK) {
    fx4 va[4], vb[4];
#pragma unroll
    for (int j = 0; j < 4; ++j) {
      int idx = tid + 256 * j;
      int r = idx >> 4, c = (idx & 15) << 2;
      va[j] = *(const fx4*)(p.A + aOff + (rowBase + r) * p.lda + k0 + c);
      vb[j] = *(const fx4*)(p.W + bOff + (long)(k0 + r) * p.ldw + colBase + c);
    }
#pragma unroll
    for (int j = 0; j < 4; ++j) {
      int idx = tid + 256 * j;
      int r = idx >> 4, c = (idx & 15) << 2;
      hx4 ha, hb;
#pragma unroll
      for (int q = 0; q < 4; ++q) {
        ha[q] = (_Float16)va[j][q];
        hb[q] = (_Float16)vb[j][q];
      }
      *(hx4*)(&As[r][c]) = ha;
      *(hx4*)(&Bs[r][c]) = hb;
    }
    __syncthreads();
    const int ar = wm * 16 + (lane & 15);
    const int aoff = (lane >> 4) * 8;
    const int n0 = wn * 32 + (lane & 15);
    const int koff = (lane >> 4) * 16;
#pragma unroll
    for (int ks = 0; ks < 2; ++ks) {
      v16h a, b0, b1;
#pragma unroll
      for (int j = 0; j < 16; ++j) {
        int kk = ks * 32 + j + aoff + ((j >= 8) ? 8 : 0);
        a[j] = As[ar][kk];
      }
#pragma unroll
      for (int j = 0; j < 16; ++j) {
        b0[j] = Bs[ks * 32 + aoff + j + ((j >= 8) ? 8 : 0)][n0];
        b1[j] = Bs[ks * 32 + aoff + j + ((j >= 8) ? 8 : 0)][n0 + 16];
      }
      acc[0] = __builtin_amdgcn_wmma_f32_16x16x32_f16(false, a, false, b0,
                                                      (short)0, acc[0], false, false);
      acc[1] = __builtin_amdgcn_wmma_f32_16x16x32_f16(false, a, false, b1,
                                                      (short)0, acc[1], false, false);
    }
    __syncthreads();
  }
  const int rsub = wm * 16 + ((lane >> 4) << 3);
  const int csub = wn * 32 + (lane & 15);
#pragma unroll
  for (int t = 0; t < 2; ++t) {
    long col = colBase + csub + t * 16;
#pragma unroll
    for (int i = 0; i < 8; ++i) {
      long row = rowBase + rsub + i;
      float v = acc[t][i];
      if (p.res) v += p.res[cOff + row * p.ldc + col];
      p.C[cOff + row * p.ldc + col] = v;
    }
  }
}

// ================= small kernels =================

// x[t] = LN(emb[ids[t]]*sqrt(D) + pe(s))
__global__ __launch_bounds__(256) void embed_ln_kernel(const int* __restrict__ ids,
                                                       const float* __restrict__ emb,
                                                       float* __restrict__ x) {
  __shared__ float red[256];
  const int t = blockIdx.x, tid = threadIdx.x;
  const int s = t % cS;
  const long tok = ids[t];
  float v[2], sum = 0.0f, sq = 0.0f;
#pragma unroll
  for (int j = 0; j < 2; ++j) {
    int d = tid + j * 256;
    float e = emb[tok * cD + d] * 22.62741699796952f;  // sqrt(512)
    float ang = (float)s * expf(-(float)(d & ~1) * (logf(10000.0f) / (float)cD));
    float pe = (d & 1) ? cosf(ang) : sinf(ang);
    v[j] = e + pe;
    sum += v[j]; sq += v[j] * v[j];
  }
  sum = block_sum(sum, red);
  sq  = block_sum(sq, red);
  float m = sum * (1.0f / cD);
  float inv = rsqrtf(sq * (1.0f / cD) - m * m + 1e-5f);
#pragma unroll
  for (int j = 0; j < 2; ++j)
    x[(long)t * cD + tid + j * 256] = (v[j] - m) * inv;
}

// out[t] = LN(in[t]); rows with sel[b] < 0 skipped when sel != nullptr
__global__ __launch_bounds__(256) void ln_kernel(float* __restrict__ out,
                                                 const float* __restrict__ in,
                                                 const int* __restrict__ sel) {
  const int t = blockIdx.x;
  if (sel && sel[t / cS] < 0) return;
  __shared__ float red[256];
  const int tid = threadIdx.x;
  float v0 = in[(long)t * cD + tid];
  float v1 = in[(long)t * cD + tid + 256];
  float sum = block_sum(v0 + v1, red);
  float sq  = block_sum(v0 * v0 + v1 * v1, red);
  float m = sum * (1.0f / cD);
  float inv = rsqrtf(sq * (1.0f / cD) - m * m + 1e-5f);
  out[(long)t * cD + tid]       = (v0 - m) * inv;
  out[(long)t * cD + tid + 256] = (v1 - m) * inv;
}

// x[t] += LN(z[t] + g[t]*ao[t]) for active rows
__global__ __launch_bounds__(256) void gate_combine_kernel(float* __restrict__ x,
    const float* __restrict__ z, const float* __restrict__ g,
    const float* __restrict__ ao, const int* __restrict__ sel) {
  const int t = blockIdx.x;
  if (sel[t / cS] < 0) return;
  __shared__ float red[256];
  const int tid = threadIdx.x;
  long i0 = (long)t * cD + tid, i1 = i0 + 256;
  float v0 = z[i0] + g[i0] * ao[i0];
  float v1 = z[i1] + g[i1] * ao[i1];
  float sum = block_sum(v0 + v1, red);
  float sq  = block_sum(v0 * v0 + v1 * v1, red);
  float m = sum * (1.0f / cD);
  float inv = rsqrtf(sq * (1.0f / cD) - m * m + 1e-5f);
  x[i0] += (v0 - m) * inv;
  x[i1] += (v1 - m) * inv;
}

__global__ __launch_bounds__(256) void add_tag_kernel(float* __restrict__ x,
    const float* __restrict__ tag, const int* __restrict__ sel) {
  const int t = blockIdx.x;
  const int e = sel[t / cS];
  if (e < 0) return;
  for (int d = threadIdx.x; d < cD; d += 256)
    x[(long)t * cD + d] += tag[(long)e * cD + d];
}

// rows of scores softmaxed with 1/sqrt(HD) scale; skip inactive batches
__global__ __launch_bounds__(256) void softmax_kernel(float* __restrict__ scores,
                                                      const int* __restrict__ sel) {
  const long row = blockIdx.x;
  if (sel[row / ((long)cNH * cS)] < 0) return;
  __shared__ float red[256];
  const int tid = threadIdx.x;
  float v = scores[row * cS + tid] * 0.125f;  // 1/sqrt(64)
  red[tid] = v; __syncthreads();
  for (int s = 128; s > 0; s >>= 1) {
    if (tid < s) red[tid] = fmaxf(red[tid], red[tid + s]);
    __syncthreads();
  }
  float mx = red[0]; __syncthreads();
  float ev = expf(v - mx);
  float sum = block_sum(ev, red);
  scores[row * cS + tid] = ev / sum;
}

__global__ __launch_bounds__(256) void mean_kernel(const float* __restrict__ x,
                                                   float* __restrict__ summary) {
  const int b = blockIdx.x;
  for (int d = threadIdx.x; d < cD; d += 256) {
    float s = 0.0f;
    for (int ss = 0; ss < cS; ++ss) s += x[((long)b * cS + ss) * cD + d];
    summary[(long)b * cD + d] = s * (1.0f / cS);
  }
}

__global__ void init_kernel(int* sel, int* visits, int* activ) {
  int t = threadIdx.x;
  if (t < cB * cE) visits[t] = 0;
  if (t < cB) { activ[t] = 1; sel[t] = -1; }
}

__global__ __launch_bounds__(256) void router_kernel(
    const float* __restrict__ summary, const float* __restrict__ r_w1,
    const float* __restrict__ r_b1, const float* __restrict__ r_w2,
    const float* __restrict__ r_b2, const float* __restrict__ r_q,
    int* __restrict__ sel, int* __restrict__ visits, int* __restrict__ activ) {
  __shared__ float hid[cB * cRH];
  __shared__ float lg[cB * (cE + 1)];
  const int tid = threadIdx.x;
  for (int idx = tid; idx < cB * cRH; idx += 256) {
    int b = idx / cRH, r = idx % cRH;
    const float* srow = summary + (long)b * cD;
    const float* wrow = r_w1 + (long)r * cD;
    float acc = r_b1[r];
    for (int k = 0; k < cD; ++k) acc += srow[k] * wrow[k];
    hid[idx] = gelu_f(acc);
  }
  __syncthreads();
  if (tid < cB * (cE + 1)) {
    int b = tid / (cE + 1), j = tid % (cE + 1);
    const float* hrow = hid + (long)b * cRH;
    const float* wrow = r_w2 + (long)j * cRH;
    float acc = r_b2[j];
    for (int k = 0; k < cRH; ++k) acc += hrow[k] * wrow[k];
    lg[tid] = acc;
  }
  __syncthreads();
  if (tid == 0) {
    for (int b = 0; b < cB; ++b) {
      float f[cE + 1];
      float m = 0.0f;
      for (int j = 0; j <= cE; ++j) { f[j] = lg[b * (cE + 1) + j] + r_q[j]; m += f[j]; }
      m *= (1.0f / (cE + 1));
      float var = 0.0f;
      for (int j = 0; j <= cE; ++j) { float d = f[j] - m; var += d * d; }
      var *= (1.0f / (cE + 1));
      float inv = rsqrtf(var + 1e-5f);
      int best = 0; float bv = -1e38f;
      for (int j = 0; j <= cE; ++j) {
        float fl = (f[j] - m) * inv;
        fl = fminf(10.0f, fmaxf(-10.0f, fl));
        if (j < cE && visits[b * cE + j] >= cMV) fl = -1e30f;
        if (fl > bv) { bv = fl; best = j; }
      }
      bool dob = (activ[b] != 0) && (best < cE);
      sel[b] = dob ? best : -1;
      if (dob) visits[b * cE + best] += 1;
      activ[b] = dob ? 1 : 0;
    }
  }
}

// ================= host launcher =================
extern "C" void kernel_launch(void* const* d_in, const int* in_sizes, int n_in,
                              void* d_out, int out_size, void* d_ws, size_t ws_size,
                              hipStream_t stream) {
  (void)in_sizes; (void)n_in; (void)out_size; (void)ws_size;
  const int*   ids    = (const int*)d_in[0];
  const float* emb    = (const float*)d_in[1];
  const float* r_w1   = (const float*)d_in[2];
  const float* r_b1   = (const float*)d_in[3];
  const float* r_w2   = (const float*)d_in[4];
  const float* r_b2   = (const float*)d_in[5];
  const float* r_q    = (const float*)d_in[6];
  const float* in_w   = (const float*)d_in[7];
  const float* in_b   = (const float*)d_in[8];
  const float* out_w  = (const float*)d_in[9];
  const float* out_b  = (const float*)d_in[10];
  const float* gate_w = (const float*)d_in[11];
  const float* gate_b = (const float*)d_in[12];
  const float* f_w1   = (const float*)d_in[13];
  const float* f_b1   = (const float*)d_in[14];
  const float* f_w2   = (const float*)d_in[15];
  const float* f_b2   = (const float*)d_in[16];
  const float* tag    = (const float*)d_in[17];
  const float* lm_w   = (const float*)d_in[18];
  const float* lm_b   = (const float*)d_in[19];
  float* out = (float*)d_out;

  // workspace layout
  float* ws      = (float*)d_ws;
  float* x       = ws;                         // T*D
  float* z       = x + (long)cT * cD;          // T*D
  float* qkv     = z + (long)cT * cD;          // T*3D
  float* scores  = qkv + (long)cT * 3 * cD;    // B*NH*S*S
  float* ao      = scores + (long)cB * cNH * cS * cS;  // T*D
  float* ao2     = ao + (long)cT * cD;         // T*D
  float* gbuf    = ao2 + (long)cT * cD;        // T*D
  float* ff1     = gbuf + (long)cT * cD;       // T*FF
  float* summary = ff1 + (long)cT * cFF;       // B*D
  int*   sel     = (int*)(summary + (long)cB * cD);
  int*   visits  = sel + cB;
  int*   activ   = visits + cB * cE;

  embed_ln_kernel<<<cT, 256, 0, stream>>>(ids, emb, x);
  init_kernel<<<1, 64, 0, stream>>>(sel, visits, activ);

  for (int step = 0; step < cMPL; ++step) {
    mean_kernel<<<cB, 256, 0, stream>>>(x, summary);
    router_kernel<<<1, 256, 0, stream>>>(summary, r_w1, r_b1, r_w2, r_b2, r_q,
                                         sel, visits, activ);
    for (int l = 0; l < cL; ++l) {
      // z = LN(x)
      ln_kernel<<<cT, 256, 0, stream>>>(z, x, sel);

      // qkv = z @ in_w[e,l]^T + in_b[e,l]
      {
        GemmP p{};
        p.A = z; p.lda = cD; p.aStrideB = (long)cS * cD; p.aStrideH = 0;
        p.W = in_w + (long)l * 3 * cD * cD; p.ldw = cD;
        p.wStrideB = 0; p.wStrideH = 0; p.wStrideSel = (long)cL * 3 * cD * cD; p.kOff = 0;
        p.bias = in_b + (long)l * 3 * cD; p.biasStrideSel = (long)cL * 3 * cD;
        p.res = nullptr;
        p.C = qkv; p.ldc = 3 * cD; p.cStrideB = (long)cS * 3 * cD; p.cStrideH = 0;
        p.K = cD; p.act = 0; p.zDiv = 1; p.sel = sel;
        gemm_wnk<<<dim3(3 * cD / BN, cS / BM, cB), 256, 0, stream>>>(p);
      }
      // scores = Q @ K^T  (batched over b,h)
      {
        GemmP p{};
        p.A = qkv; p.lda = 3 * cD; p.aStrideB = (long)cS * 3 * cD; p.aStrideH = cHD;
        p.W = qkv; p.ldw = 3 * cD;
        p.wStrideB = (long)cS * 3 * cD; p.wStrideH = cHD; p.wStrideSel = 0; p.kOff = cD;
        p.bias = nullptr; p.biasStrideSel = 0; p.res = nullptr;
        p.C = scores; p.ldc = cS;
        p.cStrideB = (long)cNH * cS * cS; p.cStrideH = (long)cS * cS;
        p.K = cHD; p.act = 0; p.zDiv = cNH; p.sel = sel;
        gemm_wnk<<<dim3(cS / BN, cS / BM, cB * cNH), 256, 0, stream>>>(p);
      }
      softmax_kernel<<<cB * cNH * cS, 256, 0, stream>>>(scores, sel);
      // ao = P @ V
      {
        GemmP p{};
        p.A = scores; p.lda = cS;
        p.aStrideB = (long)cNH * cS * cS; p.aStrideH = (long)cS * cS;
        p.W = qkv; p.ldw = 3 * cD;
        p.wStrideB = (long)cS * 3 * cD; p.wStrideH = cHD; p.wStrideSel = 0;
        p.kOff = 2 * cD;
        p.bias = nullptr; p.biasStrideSel = 0; p.res = nullptr;
        p.C = ao; p.ldc = cD; p.cStrideB = (long)cS * cD; p.cStrideH = cHD;
        p.K = cS; p.act = 0; p.zDiv = cNH; p.sel = sel;
        gemm_bkn<<<dim3(cHD / BN, cS / BM, cB * cNH), 256, 0, stream>>>(p);
      }
      // ao2 = ao @ out_w[e,l]^T + out_b[e,l]
      {
        GemmP p{};
        p.A = ao; p.lda = cD; p.aStrideB = (long)cS * cD; p.aStrideH = 0;
        p.W = out_w + (long)l * cD * cD; p.ldw = cD;
        p.wStrideB = 0; p.wStrideH = 0; p.wStrideSel = (long)cL * cD * cD; p.kOff = 0;
        p.bias = out_b + (long)l * cD; p.biasStrideSel = (long)cL * cD;
        p.res = nullptr;
        p.C = ao2; p.ldc = cD; p.cStrideB = (long)cS * cD; p.cStrideH = 0;
        p.K = cD; p.act = 0; p.zDiv = 1; p.sel = sel;
        gemm_wnk<<<dim3(cD / BN, cS / BM, cB), 256, 0, stream>>>(p);
      }
      // g = sigmoid([z, ao2] @ gate_w[e,l]^T + gate_b[e,l])  (two K-halves)
      {
        GemmP p{};
        p.A = z; p.lda = cD; p.aStrideB = (long)cS * cD; p.aStrideH = 0;
        p.W = gate_w + (long)l * cD * 2 * cD; p.ldw = 2 * cD;
        p.wStrideB = 0; p.wStrideH = 0; p.wStrideSel = (long)cL * cD * 2 * cD; p.kOff = 0;
        p.bias = gate_b + (long)l * cD; p.biasStrideSel = (long)cL * cD;
        p.res = nullptr;
        p.C = gbuf; p.ldc = cD; p.cStrideB = (long)cS * cD; p.cStrideH = 0;
        p.K = cD; p.act = 0; p.zDiv = 1; p.sel = sel;
        gemm_wnk<<<dim3(cD / BN, cS / BM, cB), 256, 0, stream>>>(p);
        // second half: accumulate + sigmoid
        p.A = ao2; p.kOff = cD; p.bias = nullptr; p.biasStrideSel = 0;
        p.res = gbuf; p.act = 2;
        gemm_wnk<<<dim3(cD / BN, cS / BM, cB), 256, 0, stream>>>(p);
      }
      // x += LN(z + g*ao2)
      gate_combine_kernel<<<cT, 256, 0, stream>>>(x, z, gbuf, ao2, sel);
      // z = LN(x)
      ln_kernel<<<cT, 256, 0, stream>>>(z, x, sel);
      // ff1 = gelu(z @ f_w1[e,l]^T + f_b1[e,l])
      {
        GemmP p{};
        p.A = z; p.lda = cD; p.aStrideB = (long)cS * cD; p.aStrideH = 0;
        p.W = f_w1 + (long)l * cFF * cD; p.ldw = cD;
        p.wStrideB = 0; p.wStrideH = 0; p.wStrideSel = (long)cL * cFF * cD; p.kOff = 0;
        p.bias = f_b1 + (long)l * cFF; p.biasStrideSel = (long)cL * cFF;
        p.res = nullptr;
        p.C = ff1; p.ldc = cFF; p.cStrideB = (long)cS * cFF; p.cStrideH = 0;
        p.K = cD; p.act = 1; p.zDiv = 1; p.sel = sel;
        gemm_wnk<<<dim3(cFF / BN, cS / BM, cB), 256, 0, stream>>>(p);
      }
      // x = x + ff1 @ f_w2[e,l]^T + f_b2[e,l]
      {
        GemmP p{};
        p.A = ff1; p.lda = cFF; p.aStrideB = (long)cS * cFF; p.aStrideH = 0;
        p.W = f_w2 + (long)l * cD * cFF; p.ldw = cFF;
        p.wStrideB = 0; p.wStrideH = 0; p.wStrideSel = (long)cL * cD * cFF; p.kOff = 0;
        p.bias = f_b2 + (long)l * cD; p.biasStrideSel = (long)cL * cD;
        p.res = x;
        p.C = x; p.ldc = cD; p.cStrideB = (long)cS * cD; p.cStrideH = 0;
        p.K = cFF; p.act = 0; p.zDiv = 1; p.sel = sel;
        gemm_wnk<<<dim3(cD / BN, cS / BM, cB), 256, 0, stream>>>(p);
      }
    }
    add_tag_kernel<<<cT, 256, 0, stream>>>(x, tag, sel);
  }

  // final LN + lm_head
  ln_kernel<<<cT, 256, 0, stream>>>(z, x, nullptr);
  {
    GemmP p{};
    p.A = z; p.lda = cD; p.aStrideB = 0; p.aStrideH = 0;
    p.W = lm_w; p.ldw = cD;
    p.wStrideB = 0; p.wStrideH = 0; p.wStrideSel = 0; p.kOff = 0;
    p.bias = lm_b; p.biasStrideSel = 0;
    p.res = nullptr;
    p.C = out; p.ldc = cV; p.cStrideB = 0; p.cStrideH = 0;
    p.K = cD; p.act = 0; p.zDiv = 1; p.sel = nullptr;
    gemm_wnk<<<dim3(cV / BN, cT / BM, 1), 256, 0, stream>>>(p);
  }
}
